// SchNet_42099269435467
// MI455X (gfx1250) — matrix operand order repaired and checked by reading, not compiled, not run on hardware
//
#include <hip/hip_runtime.h>
#include <hip/hip_bf16.h>
#include <math.h>

typedef __attribute__((ext_vector_type(16))) _Float16 v16h;
typedef __attribute__((ext_vector_type(8)))  float    v8f;

#define N_ATOMS 50000
#define N_PAD   50048          // 391 * 128: removes all row-bound checks
#define E_EDGES 800000
#define HDIM 64
#define FDIM 64
#define GDIM 50
#define LAYERS 6
#define BGRAPH 128
#define CUTOFF_R 10.0f
#define PI_F 3.14159265358979f

// ---- CDNA5 WMMA fragment index maps (wave32, 16x16x32 f16) -----------------
// A (16x32, lane holds row m = lane&15):  k(j) = (j>>3)*16 + (lane>>4)*8 + (j&7)
// B (32x16, lane holds col n = lane&15):  k(j) = (j>>3)*16 + (lane>>4)*8 + (j&7)
// C (16x16 f32): lane -> n = lane&15 ; reg r -> m = r + (lane>>4)*8
__device__ __forceinline__ int a_kidx(int lane, int j) {
    return ((j >> 3) << 4) + ((lane >> 4) << 3) + (j & 7);
}
__device__ __forceinline__ int b_kidx(int lane, int j) {
    return ((lane >> 4) << 4) + j;      // j-contiguous: 32B run in a [n][k] layout
}
__device__ __forceinline__ v8f wmma16(v16h a, v16h b, v8f c) {
    return __builtin_amdgcn_wmma_f32_16x16x32_f16(false, a, false, b, (short)0, c, false, false);
}

// Build a B fragment from LDS weights stored TRANSPOSED as sWt[n*64 + k];
// two contiguous 8-half runs (k = 8h.., 16+8h..) -> lowers to 2x ds_load_b128.
__device__ __forceinline__ v16h bfrag(const _Float16* sWt, int lane, int kk, int ncol) {
    v16h b;
    const _Float16* p = sWt + ncol * 64 + kk * 32 + ((lane >> 4) << 3);
#pragma unroll
    for (int j = 0; j < 16; ++j) b[j] = p[((j >> 3) << 4) + (j & 7)];
    return b;
}

// ---- edge geometry: d[e], cosine cutoff[e] ---------------------------------
__global__ void k_edge_geom(const float* __restrict__ pos, const int* __restrict__ ei,
                            float* __restrict__ d, float* __restrict__ cut) {
    int e = blockIdx.x * blockDim.x + threadIdx.x;
    if (e >= E_EDGES) return;
    int s = ei[e], t = ei[E_EDGES + e];
    float dx = pos[s * 3 + 0] - pos[t * 3 + 0];
    float dy = pos[s * 3 + 1] - pos[t * 3 + 1];
    float dz = pos[s * 3 + 2] - pos[t * 3 + 2];
    float dd = sqrtf(dx * dx + dy * dy + dz * dz);
    d[e] = dd;
    float c = 0.5f * (__cosf(dd * (PI_F / CUTOFF_R)) + 1.0f);
    cut[e] = (dd < CUTOFF_R) ? c : 0.0f;
}

// ---- initial node features: h = embed[z] -----------------------------------
__global__ void k_embed(const int* __restrict__ z, const float* __restrict__ embed,
                        float* __restrict__ h) {
    int t = blockIdx.x * blockDim.x + threadIdx.x;
    if (t >= N_ATOMS * HDIM) return;
    int i = t >> 6, c = t & 63;
    h[t] = embed[z[i] * HDIM + c];
}

// ---- h_j = h @ lin_w[l]  (WMMA), also zeroes agg rows ----------------------
__global__ void __launch_bounds__(256) k_hj(const float* __restrict__ h,
                                            const float* __restrict__ W,
                                            float* __restrict__ hj,
                                            float* __restrict__ agg) {
    __shared__ _Float16 sWt[64 * 64];   // transposed: sWt[n*64 + k]
    __shared__ _Float16 sA[8][16 * 64];
    int tid = threadIdx.x, lane = tid & 31, wv = tid >> 5;
    for (int i = tid; i < 64 * 64; i += 256) {
        int k = i >> 6, n = i & 63;
        sWt[n * 64 + k] = (_Float16)W[i];
    }
    __syncthreads();

    int n0 = lane & 15, m0 = lane & 15, hi8 = (lane >> 4) << 3;
    int base = blockIdx.x * 128 + wv * 16;   // base+16 <= N_PAD always

    _Float16* T = sA[wv];
    for (int idx = lane; idx < 16 * 64; idx += 32) {
        int off = (base + (idx >> 6)) * 64 + (idx & 63);
        T[idx] = (_Float16)h[off];
        agg[off] = 0.0f;
    }

    v16h af[2];
#pragma unroll
    for (int kk = 0; kk < 2; ++kk)
#pragma unroll
        for (int j = 0; j < 16; ++j)
            af[kk][j] = T[m0 * 64 + kk * 32 + a_kidx(lane, j)];

#pragma unroll
    for (int nt = 0; nt < 4; ++nt) {
        v8f c = {0.f, 0.f, 0.f, 0.f, 0.f, 0.f, 0.f, 0.f};
        c = wmma16(af[0], bfrag(sWt, lane, 0, nt * 16 + n0), c);
        c = wmma16(af[1], bfrag(sWt, lane, 1, nt * 16 + n0), c);
        int f = nt * 16 + n0;
#pragma unroll
        for (int r = 0; r < 8; ++r)
            hj[(base + r + hi8) * 64 + f] = c[r];
    }
}

// ---- h = h + (agg @ lin_out_w[l] + b)  (WMMA) ------------------------------
__global__ void __launch_bounds__(256) k_out(const float* __restrict__ agg,
                                             const float* __restrict__ W,
                                             const float* __restrict__ bias,
                                             const float* __restrict__ hin,
                                             float* __restrict__ hout) {
    __shared__ _Float16 sWt[64 * 64];   // transposed
    __shared__ _Float16 sA[8][16 * 64];
    int tid = threadIdx.x, lane = tid & 31, wv = tid >> 5;
    for (int i = tid; i < 64 * 64; i += 256) {
        int k = i >> 6, n = i & 63;
        sWt[n * 64 + k] = (_Float16)W[i];
    }
    __syncthreads();

    int n0 = lane & 15, m0 = lane & 15, hi8 = (lane >> 4) << 3;
    int base = blockIdx.x * 128 + wv * 16;

    _Float16* T = sA[wv];
    for (int idx = lane; idx < 16 * 64; idx += 32)
        T[idx] = (_Float16)agg[(base + (idx >> 6)) * 64 + (idx & 63)];

    v16h af[2];
#pragma unroll
    for (int kk = 0; kk < 2; ++kk)
#pragma unroll
        for (int j = 0; j < 16; ++j)
            af[kk][j] = T[m0 * 64 + kk * 32 + a_kidx(lane, j)];

#pragma unroll
    for (int nt = 0; nt < 4; ++nt) {
        v8f c = {0.f, 0.f, 0.f, 0.f, 0.f, 0.f, 0.f, 0.f};
        c = wmma16(af[0], bfrag(sWt, lane, 0, nt * 16 + n0), c);
        c = wmma16(af[1], bfrag(sWt, lane, 1, nt * 16 + n0), c);
        int f = nt * 16 + n0;
        float bb = bias[f];
#pragma unroll
        for (int r = 0; r < 8; ++r) {
            int row = base + r + hi8;
            hout[row * 64 + f] = hin[row * 64 + f] + c[r] + bb;
        }
    }
}

// ---- fused per-edge filter MLP (2x WMMA GEMM) + message scatter ------------
__global__ void __launch_bounds__(128) k_edge(const float* __restrict__ d,
                                              const float* __restrict__ cut,
                                              const int* __restrict__ ei,
                                              const float* __restrict__ W1,
                                              const float* __restrict__ b1,
                                              const float* __restrict__ W2,
                                              const float* __restrict__ b2,
                                              const float* __restrict__ hj,
                                              float* __restrict__ agg) {
    __shared__ _Float16 sW1t[64 * 64];  // transposed, G padded 50 -> 64
    __shared__ _Float16 sW2t[64 * 64];  // transposed
    __shared__ _Float16 sT[4][16 * 64]; // tanh() intermediate per wave
    __shared__ __align__(16) float sHj[4][16 * 64];
    __shared__ int   sDst[4][16];
    __shared__ float sCut[4][16];

    int tid = threadIdx.x, lane = tid & 31, wv = tid >> 5;
    for (int i = tid; i < 64 * 64; i += 128) {
        int k = i >> 6, n = i & 63;
        sW1t[n * 64 + k] = (k < GDIM) ? (_Float16)W1[k * 64 + n] : (_Float16)0.0f;
        sW2t[n * 64 + k] = (_Float16)W2[i];
    }
    __syncthreads();

    int n0 = lane & 15, hi8 = (lane >> 4) << 3;
    int tileBase = blockIdx.x * 64 + wv * 16;  // E % 64 == 0: every tile full

    // stage dst / cut / gathered h_j[src] rows for this wave's 16 edges
    const float4* hj4 = (const float4*)hj;
    float4* sH4 = (float4*)sHj[wv];
    for (int idx = lane; idx < 16 * 16; idx += 32) {
        int r = idx >> 4, c4 = idx & 15;
        int e = tileBase + r;
        int src = ei[e];
        sH4[r * 16 + c4] = hj4[src * 16 + c4];
        if (c4 == 0) { sDst[wv][r] = ei[E_EDGES + e]; sCut[wv][r] = cut[e]; }
    }

    // Gaussian smearing straight into A-fragment register layout
    float dm = d[tileBase + (lane & 15)];
    const float step  = CUTOFF_R / (float)(GDIM - 1);
    const float coeff = -0.5f / (step * step);
    v16h af[2];
#pragma unroll
    for (int kk = 0; kk < 2; ++kk)
#pragma unroll
        for (int j = 0; j < 16; ++j) {
            int k = kk * 32 + a_kidx(lane, j);
            float v = 0.0f;
            if (k < GDIM) { float t = dm - step * (float)k; v = __expf(coeff * t * t); }
            af[kk][j] = (_Float16)v;
        }

    // GEMM1: tanh(attr @ W1 + b1) -> LDS (transposes C layout back to A layout)
    _Float16* T = sT[wv];
#pragma unroll
    for (int nt = 0; nt < 4; ++nt) {
        v8f c = {0.f, 0.f, 0.f, 0.f, 0.f, 0.f, 0.f, 0.f};
        c = wmma16(af[0], bfrag(sW1t, lane, 0, nt * 16 + n0), c);
        c = wmma16(af[1], bfrag(sW1t, lane, 1, nt * 16 + n0), c);
        float bb = b1[nt * 16 + n0];
#pragma unroll
        for (int r = 0; r < 8; ++r)
            T[(r + hi8) * 64 + nt * 16 + n0] = (_Float16)tanhf(c[r] + bb);
    }

    // GEMM2: W = T @ W2 + b2 ; then msg = h_j[src] * W * cut -> atomic scatter
    v16h a2[2];
#pragma unroll
    for (int kk = 0; kk < 2; ++kk)
#pragma unroll
        for (int j = 0; j < 16; ++j)
            a2[kk][j] = T[(lane & 15) * 64 + kk * 32 + a_kidx(lane, j)];

#pragma unroll
    for (int nt = 0; nt < 4; ++nt) {
        v8f c = {0.f, 0.f, 0.f, 0.f, 0.f, 0.f, 0.f, 0.f};
        c = wmma16(a2[0], bfrag(sW2t, lane, 0, nt * 16 + n0), c);
        c = wmma16(a2[1], bfrag(sW2t, lane, 1, nt * 16 + n0), c);
        int f = nt * 16 + n0;
        float bb = b2[f];
#pragma unroll
        for (int r = 0; r < 8; ++r) {
            int m = r + hi8;
            float wf  = (c[r] + bb) * sCut[wv][m];
            float val = sHj[wv][m * 64 + f] * wf;
            atomicAdd(&agg[sDst[wv][m] * 64 + f], val);
        }
    }
}

// ---- global mean pool + head -----------------------------------------------
__global__ void k_zero(float* __restrict__ p, int n) {
    int t = blockIdx.x * blockDim.x + threadIdx.x;
    if (t < n) p[t] = 0.0f;
}

__global__ void k_pool(const float* __restrict__ h, const int* __restrict__ batch,
                       float* __restrict__ pool, float* __restrict__ cnt) {
    int t = blockIdx.x * blockDim.x + threadIdx.x;
    if (t >= N_ATOMS * 64) return;
    int i = t >> 6, c = t & 63;
    int b = batch[i];
    atomicAdd(&pool[b * 64 + c], h[t]);
    if (c == 0) atomicAdd(&cnt[b], 1.0f);
}

__global__ void k_head(const float* __restrict__ pool, const float* __restrict__ cnt,
                       const float* __restrict__ w1, const float* __restrict__ bb1,
                       const float* __restrict__ w2, const float* __restrict__ bb2,
                       float* __restrict__ out) {
    int g = threadIdx.x;
    if (g >= BGRAPH) return;
    float inv = 1.0f / fmaxf(cnt[g], 1.0f);
    float p[64];
#pragma unroll
    for (int i = 0; i < 64; ++i) p[i] = pool[g * 64 + i] * inv;
    float acc = bb2[0];
    for (int j = 0; j < 32; ++j) {
        float s = bb1[j];
#pragma unroll
        for (int i = 0; i < 64; ++i) s += p[i] * w1[i * 32 + j];
        acc += fmaxf(s, 0.0f) * w2[j];
    }
    out[g] = acc;
}

extern "C" void kernel_launch(void* const* d_in, const int* in_sizes, int n_in,
                              void* d_out, int out_size, void* d_ws, size_t ws_size,
                              hipStream_t stream) {
    (void)in_sizes; (void)n_in; (void)out_size; (void)ws_size;
    const int*   z      = (const int*)d_in[0];
    const float* pos    = (const float*)d_in[1];
    const int*   batch  = (const int*)d_in[2];
    const int*   ei     = (const int*)d_in[3];
    const float* embed  = (const float*)d_in[4];
    const float* mlp_w1 = (const float*)d_in[5];
    const float* mlp_b1 = (const float*)d_in[6];
    const float* mlp_w2 = (const float*)d_in[7];
    const float* mlp_b2 = (const float*)d_in[8];
    const float* lin_w  = (const float*)d_in[9];
    const float* low    = (const float*)d_in[10];
    const float* lob    = (const float*)d_in[11];
    const float* fc1w   = (const float*)d_in[12];
    const float* fc1b   = (const float*)d_in[13];
    const float* fc2w   = (const float*)d_in[14];
    const float* fc2b   = (const float*)d_in[15];
    float* out = (float*)d_out;

    char* ws = (char*)d_ws;
    size_t off = 0;
    auto alloc = [&](size_t nfloat) {
        float* p = (float*)(ws + off);
        off += nfloat * sizeof(float);
        return p;
    };
    float* d    = alloc(E_EDGES);
    float* cut  = alloc(E_EDGES);
    float* hA   = alloc((size_t)N_PAD * 64);
    float* hB   = alloc((size_t)N_PAD * 64);
    float* hj   = alloc((size_t)N_PAD * 64);
    float* agg  = alloc((size_t)N_PAD * 64);
    float* pool = alloc(BGRAPH * 64);
    float* cnt  = alloc(BGRAPH);   // contiguous after pool (zeroed together)

    k_edge_geom<<<(E_EDGES + 255) / 256, 256, 0, stream>>>(pos, ei, d, cut);
    k_embed<<<(N_ATOMS * 64 + 255) / 256, 256, 0, stream>>>(z, embed, hA);

    float* hcur = hA; float* hnxt = hB;
    for (int l = 0; l < LAYERS; ++l) {
        k_hj<<<N_PAD / 128, 256, 0, stream>>>(hcur, lin_w + (size_t)l * 64 * 64, hj, agg);
        k_edge<<<E_EDGES / 64, 128, 0, stream>>>(d, cut, ei,
            mlp_w1 + (size_t)l * GDIM * 64, mlp_b1 + (size_t)l * 64,
            mlp_w2 + (size_t)l * 64 * 64,   mlp_b2 + (size_t)l * 64, hj, agg);
        k_out<<<N_PAD / 128, 256, 0, stream>>>(agg, low + (size_t)l * 64 * 64,
                                               lob + (size_t)l * 64, hcur, hnxt);
        float* t = hcur; hcur = hnxt; hnxt = t;
    }

    k_zero<<<(BGRAPH * 64 + BGRAPH + 255) / 256, 256, 0, stream>>>(pool, BGRAPH * 64 + BGRAPH);
    k_pool<<<(N_ATOMS * 64 + 255) / 256, 256, 0, stream>>>(hcur, batch, pool, cnt);
    k_head<<<1, 128, 0, stream>>>(pool, cnt, fc1w, fc1b, fc2w, fc2b, out);
}
